// FPMatMulMESO_8194797601024
// MI455X (gfx1250) — 8-bit matrix operand order repaired and checked by reading, compiled, not run on hardware
//
#include <hip/hip_runtime.h>

typedef int v8i __attribute__((ext_vector_type(8)));

static constexpr int Bb = 32, Mm = 2048, Nn = 2048, Kk = 512;
static_assert(Kk == 512, "pipeline below assumes exactly two 256-wide K chunks");

static constexpr size_t PLANE = (size_t)Bb * Mm * Kk;
static constexpr size_t OFF_SUMW = 256;
static constexpr size_t OFF_SUMB = OFF_SUMW + (size_t)Bb * Mm * 4;
static constexpr size_t OFF_WLO  = OFF_SUMB + (size_t)Bb * Nn * 4;
static constexpr size_t OFF_WHI  = OFF_WLO + PLANE;
static constexpr size_t OFF_BLO  = OFF_WHI + PLANE;
static constexpr size_t OFF_BHI  = OFF_BLO + PLANE;

#if defined(__has_builtin)
#if __has_builtin(__builtin_amdgcn_global_load_async_to_lds_b128) && \
    __has_builtin(__builtin_amdgcn_s_wait_asynccnt)
#define USE_ASYNC_LDS 1
#endif
#endif
#ifndef USE_ASYNC_LDS
#define USE_ASYNC_LDS 0
#endif

#if USE_ASYNC_LDS
typedef int v4i_g __attribute__((vector_size(16)));
typedef v4i_g __attribute__((address_space(1)))* g_v4i_p;
typedef v4i_g __attribute__((address_space(3)))* l_v4i_p;
#endif

__device__ inline void wait_async_all() {
#if USE_ASYNC_LDS
  __builtin_amdgcn_s_wait_asynccnt(0);
#endif
}

__device__ inline unsigned encF(float f) {
  unsigned u = __float_as_uint(f);
  return (u & 0x80000000u) ? ~u : (u | 0x80000000u);
}
__device__ inline float decF(unsigned e) {
  return (e & 0x80000000u) ? __uint_as_float(e & 0x7fffffffu) : __uint_as_float(~e);
}

__global__ void init_header_kernel(unsigned* hu) {
  if (threadIdx.x == 0) {
    hu[0] = 0xFFFFFFFFu; hu[1] = 0u;
    hu[2] = 0xFFFFFFFFu; hu[3] = 0u;
  }
}

__global__ void minmax_kernel(const float* __restrict__ x, size_t n,
                              unsigned* encMin, unsigned* encMax) {
  __shared__ float smin[8], smax[8];
  float lmin = 3.402823466e38f, lmax = -3.402823466e38f;
  size_t stride = (size_t)gridDim.x * blockDim.x;
  for (size_t i = (size_t)blockIdx.x * blockDim.x + threadIdx.x; i < n; i += stride) {
    float v = x[i];
    lmin = fminf(lmin, v);
    lmax = fmaxf(lmax, v);
  }
  for (int off = 16; off; off >>= 1) {
    lmin = fminf(lmin, __shfl_xor(lmin, off, 32));
    lmax = fmaxf(lmax, __shfl_xor(lmax, off, 32));
  }
  int lane = threadIdx.x & 31, w = threadIdx.x >> 5;
  if (lane == 0) { smin[w] = lmin; smax[w] = lmax; }
  __syncthreads();
  if (threadIdx.x == 0) {
    int nw = blockDim.x >> 5;
    for (int i = 1; i < nw; ++i) { lmin = fminf(lmin, smin[i]); lmax = fmaxf(lmax, smax[i]); }
    atomicMin(encMin, encF(lmin));
    atomicMax(encMax, encF(lmax));
  }
}

__global__ void finalize_kernel(float* hf, const unsigned* hu, const int* bwPtr) {
  if (threadIdx.x != 0) return;
  int bw = bwPtr[0];
  float qmax = (float)((1u << bw) - 1u);
  {
    float mn = decF(hu[0]), mx = decF(hu[1]);
    float maxval = fmaxf(fmaxf(mx - mn, fabsf(mn)), 1e-12f);
    float shift = floorf(log2f(qmax / maxval));
    float scale = exp2f(shift);
    hf[8] = mn; hf[9] = scale; hf[10] = roundf(mn * scale); hf[11] = shift;
  }
  {
    float mn = decF(hu[2]), mx = decF(hu[3]);
    float maxval = fmaxf(fmaxf(mx - mn, fabsf(mn)), 1e-12f);
    float shift = floorf(log2f(qmax / maxval));
    float scale = exp2f(shift);
    hf[12] = mn; hf[13] = scale; hf[14] = roundf(mn * scale); hf[15] = shift;
  }
  hf[16] = exp2f(-(hf[11] + hf[15]));
}

__global__ void quantize_kernel(const float4* __restrict__ x,
                                unsigned* __restrict__ qlo, unsigned* __restrict__ qhi,
                                size_t n4, const float* __restrict__ hf, int base) {
  float mn = hf[base], scale = hf[base + 1];
  size_t stride = (size_t)gridDim.x * blockDim.x;
  for (size_t i = (size_t)blockIdx.x * blockDim.x + threadIdx.x; i < n4; i += stride) {
    float4 v = x[i];
    unsigned q0 = (unsigned)fminf(roundf((v.x - mn) * scale), 65535.0f);
    unsigned q1 = (unsigned)fminf(roundf((v.y - mn) * scale), 65535.0f);
    unsigned q2 = (unsigned)fminf(roundf((v.z - mn) * scale), 65535.0f);
    unsigned q3 = (unsigned)fminf(roundf((v.w - mn) * scale), 65535.0f);
    qlo[i] = (q0 & 0xFFu) | ((q1 & 0xFFu) << 8) | ((q2 & 0xFFu) << 16) | ((q3 & 0xFFu) << 24);
    qhi[i] = (q0 >> 8) | ((q1 >> 8) << 8) | ((q2 >> 8) << 16) | ((q3 >> 8) << 24);
  }
}

__device__ inline unsigned sum4b(unsigned a) {
  return (a & 0xFFu) + ((a >> 8) & 0xFFu) + ((a >> 16) & 0xFFu) + (a >> 24);
}

__global__ void rowsum_kernel(const unsigned char* __restrict__ lo,
                              const unsigned char* __restrict__ hi,
                              float* __restrict__ sums, int rows) {
  int wave = blockIdx.x * (blockDim.x >> 5) + (threadIdx.x >> 5);
  if (wave >= rows) return;
  int lane = threadIdx.x & 31;
  uint4 vl = *(const uint4*)(lo + (size_t)wave * Kk + lane * 16);
  uint4 vh = *(const uint4*)(hi + (size_t)wave * Kk + lane * 16);
  unsigned s = sum4b(vl.x) + sum4b(vl.y) + sum4b(vl.z) + sum4b(vl.w);
  unsigned sh = sum4b(vh.x) + sum4b(vh.y) + sum4b(vh.z) + sum4b(vh.w);
  unsigned tot = s + (sh << 8);
  for (int off = 16; off; off >>= 1) tot += __shfl_xor(tot, off, 32);
  if (lane == 0) sums[wave] = (float)tot;
}

__device__ inline v8i fragA(const unsigned char* plane, int row, int kc, int lane) {
  const unsigned char* rp = plane + (size_t)(row + (lane & 15)) * 256 + kc + ((lane >> 4) * 16);
  v8i f;
#pragma unroll
  for (int g = 0; g < 4; ++g) {
    uint2 d = *(const uint2*)(rp + (g >> 1) * 32 + (g & 1) * 8);
    f[2 * g] = (int)d.x;
    f[2 * g + 1] = (int)d.y;
  }
  return f;
}

__device__ inline v8i fragB(const unsigned char* plane, int col, int kc, int lane) {
  const unsigned char* rp = plane + (size_t)(col + (lane & 15)) * 256 + kc + ((lane >> 4) * 16);
  uint4 d0 = *(const uint4*)(rp);
  uint4 d1 = *(const uint4*)(rp + 32);
  v8i f;
  f[0] = (int)d0.x; f[1] = (int)d0.y; f[2] = (int)d0.z; f[3] = (int)d0.w;
  f[4] = (int)d1.x; f[5] = (int)d1.y; f[6] = (int)d1.z; f[7] = (int)d1.w;
  return f;
}

__device__ inline void stage_plane(const unsigned char* __restrict__ gsrc, unsigned char* lds,
                                   int rows, int kk, int tid) {
  int total = rows * 16;
  for (int idx = tid; idx < total; idx += 256) {
    int row = idx >> 4;
    int col = (idx & 15) * 16;
    const unsigned char* s = gsrc + (size_t)row * Kk + kk * 256 + col;
    unsigned char* d = lds + row * 256 + col;
#if USE_ASYNC_LDS
    __builtin_amdgcn_global_load_async_to_lds_b128((g_v4i_p)(void*)s, (l_v4i_p)(void*)d, 0, 0);
#else
    *(uint4*)d = *(const uint4*)s;
#endif
  }
}

__global__ __launch_bounds__(256) void meso_gemm_kernel(
    const unsigned char* __restrict__ Wlo, const unsigned char* __restrict__ Whi,
    const unsigned char* __restrict__ Blo, const unsigned char* __restrict__ Bhi,
    const float* __restrict__ sumW, const float* __restrict__ sumB,
    const float* __restrict__ hf, float* __restrict__ out) {
  extern __shared__ unsigned char smem[];

  const int tid = threadIdx.x;
  const int lane = tid & 31;
  const int waveId = tid >> 5;
  const int waveM = (waveId >> 1) * 32;
  const int waveN = (waveId & 1) * 32;
  const int batch = blockIdx.z;
  const int blockRow = blockIdx.y * 128;
  const int blockCol = blockIdx.x * 64;

  const size_t wOff = ((size_t)batch * Mm + blockRow) * Kk;
  const size_t bOff = ((size_t)batch * Nn + blockCol) * Kk;
  const unsigned char* wLo = Wlo + wOff;
  const unsigned char* wHi = Whi + wOff;
  const unsigned char* bLo = Blo + bOff;
  const unsigned char* bHi = Bhi + bOff;

  const v8i zero = {0, 0, 0, 0, 0, 0, 0, 0};
  v8i acc_hh[2][2], acc_mid[2][2], acc_ll[2][2];
#pragma unroll
  for (int rb = 0; rb < 2; ++rb)
#pragma unroll
    for (int cb = 0; cb < 2; ++cb) {
      acc_hh[rb][cb] = zero;
      acc_mid[rb][cb] = zero;
      acc_ll[rb][cb] = zero;
    }

  auto stage_chunk = [&](int buf, int kk) {
    unsigned char* base = smem + buf * 98304;
    stage_plane(wLo, base, 128, kk, tid);
    stage_plane(wHi, base + 32768, 128, kk, tid);
    stage_plane(bLo, base + 65536, 64, kk, tid);
    stage_plane(bHi, base + 81920, 64, kk, tid);
  };

  auto compute_chunk = [&](int buf) {
    unsigned char* aLoP = smem + buf * 98304;
    unsigned char* aHiP = aLoP + 32768;
    unsigned char* bLoP = aLoP + 65536;
    unsigned char* bHiP = aLoP + 81920;
#pragma unroll
    for (int kc = 0; kc < 256; kc += 64) {
      v8i aL[2], aH[2], bL[2], bH[2];
#pragma unroll
      for (int rb = 0; rb < 2; ++rb) {
        aL[rb] = fragA(aLoP, waveM + rb * 16, kc, lane);
        aH[rb] = fragA(aHiP, waveM + rb * 16, kc, lane);
      }
#pragma unroll
      for (int cb = 0; cb < 2; ++cb) {
        bL[cb] = fragB(bLoP, waveN + cb * 16, kc, lane);
        bH[cb] = fragB(bHiP, waveN + cb * 16, kc, lane);
      }
#pragma unroll
      for (int rb = 0; rb < 2; ++rb)
#pragma unroll
        for (int cb = 0; cb < 2; ++cb) {
          acc_hh[rb][cb] = __builtin_amdgcn_wmma_i32_16x16x64_iu8(
              false, aH[rb], false, bH[cb], acc_hh[rb][cb], false, false);
          acc_mid[rb][cb] = __builtin_amdgcn_wmma_i32_16x16x64_iu8(
              false, aH[rb], false, bL[cb], acc_mid[rb][cb], false, false);
          acc_mid[rb][cb] = __builtin_amdgcn_wmma_i32_16x16x64_iu8(
              false, aL[rb], false, bH[cb], acc_mid[rb][cb], false, false);
          acc_ll[rb][cb] = __builtin_amdgcn_wmma_i32_16x16x64_iu8(
              false, aL[rb], false, bL[cb], acc_ll[rb][cb], false, false);
        }
    }
  };

  stage_chunk(0, 0);
  wait_async_all();
  __syncthreads();
  stage_chunk(1, 1);
  compute_chunk(0);
  wait_async_all();
  __syncthreads();
  compute_chunk(1);

  const float minWq = hf[10], minBq = hf[14], invOut = hf[16];
  const float cterm = (float)Kk * minWq * minBq;
  const int h = lane >> 4, nn = lane & 15;
#pragma unroll
  for (int rb = 0; rb < 2; ++rb)
#pragma unroll
    for (int cb = 0; cb < 2; ++cb) {
      int nIdx = blockCol + waveN + cb * 16 + nn;
      float sB = sumB[(size_t)batch * Nn + nIdx];
      int mBase = blockRow + waveM + rb * 16 + 8 * h;
#pragma unroll
      for (int r = 0; r < 8; ++r) {
        int m = mBase + r;
        float sW = sumW[(size_t)batch * Mm + m];
        float v = 65536.0f * (float)acc_hh[rb][cb][r] +
                  256.0f * (float)acc_mid[rb][cb][r] + (float)acc_ll[rb][cb][r];
        v = (v + minBq * sW + minWq * sB + cterm) * invOut;
        out[((size_t)batch * Mm + m) * Nn + nIdx] = v;
      }
    }
}

extern "C" void kernel_launch(void* const* d_in, const int* in_sizes, int n_in,
                              void* d_out, int out_size, void* d_ws, size_t ws_size,
                              hipStream_t stream) {
  (void)in_sizes; (void)n_in; (void)out_size; (void)ws_size;
  const float* Q = (const float*)d_in[0];
  const float* Keys = (const float*)d_in[1];
  const int* bw = (const int*)d_in[2];
  float* out = (float*)d_out;
  unsigned char* ws = (unsigned char*)d_ws;

  unsigned* hu = (unsigned*)ws;
  float* hf = (float*)ws;
  float* sumW = (float*)(ws + OFF_SUMW);
  float* sumB = (float*)(ws + OFF_SUMB);
  unsigned char* Wlo = ws + OFF_WLO;
  unsigned char* Whi = ws + OFF_WHI;
  unsigned char* Blo = ws + OFF_BLO;
  unsigned char* Bhi = ws + OFF_BHI;

  const size_t nW = (size_t)Bb * Mm * Kk;
  const size_t nB = (size_t)Bb * Nn * Kk;

  init_header_kernel<<<1, 32, 0, stream>>>(hu);
  minmax_kernel<<<1024, 256, 0, stream>>>(Q, nW, hu + 0, hu + 1);
  minmax_kernel<<<1024, 256, 0, stream>>>(Keys, nB, hu + 2, hu + 3);
  finalize_kernel<<<1, 32, 0, stream>>>(hf, hu, bw);
  quantize_kernel<<<4096, 256, 0, stream>>>((const float4*)Q, (unsigned*)Wlo, (unsigned*)Whi,
                                            nW / 4, hf, 8);
  quantize_kernel<<<4096, 256, 0, stream>>>((const float4*)Keys, (unsigned*)Blo, (unsigned*)Bhi,
                                            nB / 4, hf, 12);
  rowsum_kernel<<<(Bb * Mm) / 8, 256, 0, stream>>>(Wlo, Whi, sumW, Bb * Mm);
  rowsum_kernel<<<(Bb * Nn) / 8, 256, 0, stream>>>(Blo, Bhi, sumB, Bb * Nn);
  meso_gemm_kernel<<<dim3(Nn / 64, Mm / 128, Bb), 256, 2 * 98304, stream>>>(
      Wlo, Whi, Blo, Bhi, sumW, sumB, hf, out);
}
